// MultiHeadedAttention_88579405512935
// MI455X (gfx1250) — matrix operand order repaired and checked by reading, not compiled, not run on hardware
//
#include <hip/hip_runtime.h>

// Problem constants (reference: B=4, T=2048, d=1024, H=16, hd=64)
#define B_   4
#define T_   2048
#define D_   1024
#define H_   16
#define HD_  64
#define D3_  3072
#define SCALE_ 0.125f   // 64^-0.5

typedef __attribute__((ext_vector_type(16))) _Float16 v16h;
typedef __attribute__((ext_vector_type(8)))  _Float16 v8h;
typedef __attribute__((ext_vector_type(4)))  _Float16 v4h;
typedef __attribute__((ext_vector_type(8)))  float    v8f;
typedef __attribute__((ext_vector_type(4)))  unsigned int tdm_v4u;
typedef __attribute__((ext_vector_type(8)))  int          tdm_v8i;
typedef __attribute__((ext_vector_type(4)))  int          tdm_v4i;

// A-matrix (16x32 f16) element->K mapping (contiguous, matches B fills):
//   half 0 lanes: K = 0..15
//   half 1 lanes: K = 16..31
__device__ __forceinline__ int a_k(int e, int hf) {
  return e + 16 * hf;
}

// Load a 16x32 A fragment for row `arow` (f16 row-major), K-chunk at k0:
// K = k0 + 16*hf .. +15 as two 16-byte runs per lane.
__device__ __forceinline__ v16h load_a_frag(const _Float16* __restrict__ arow,
                                            int k0, int hf) {
  v8h lo = *(const v8h*)(arow + k0 + 16 * hf);
  v8h hi = *(const v8h*)(arow + k0 + 16 * hf + 8);
  v16h a;
#pragma unroll
  for (int e = 0; e < 8; ++e) { a[e] = lo[e]; a[e + 8] = hi[e]; }
  return a;
}

// ---------------------------------------------------------------------------
// TDM: DMA one 64-row x 32-half f16 tile (row stride 1024 halves) from global
// into LDS at lds_byte_addr. D# per CDNA5 ISA ch.8: tensor dims == tile dims
// (no OOB path), data_size=2B, 2-D tensor (groups 2/3 zero / unused).
// Issue from ONE wave; completion tracked with TENSORcnt.
// (6-arg builtin variant: g0, g1, g2, g3, extra group, cpol)
// ---------------------------------------------------------------------------
__device__ __forceinline__ void tdm_load_tile64x32(const _Float16* gsrc,
                                                   unsigned lds_byte_addr) {
  unsigned long long ga = (unsigned long long)(uintptr_t)gsrc;
  tdm_v4u g0;
  g0[0] = 1u;                                        // count=1 (valid user D#)
  g0[1] = lds_byte_addr;                             // lds_addr
  g0[2] = (unsigned)(ga & 0xffffffffu);              // global_addr[31:0]
  g0[3] = (unsigned)((ga >> 32) & 0x01ffffffu)       // global_addr[56:32]
          | (2u << 30);                              // type=2 ("image")
  tdm_v8i g1;
  g1[0] = 0x00010000;                                // data_size=1 (2 bytes)
  g1[1] = (int)(32u << 16);                          // tensor_dim0 = 32
  g1[2] = (int)(64u << 16);                          // dim0_hi=0 | tensor_dim1 = 64
  g1[3] = (int)(32u << 16);                          // dim1_hi=0 | tile_dim0 = 32
  g1[4] = 64;                                        // tile_dim1 = 64 (tile_dim2=0)
  g1[5] = 1024;                                      // tensor_dim0_stride
  g1[6] = 0;
  g1[7] = 0;
  tdm_v4i z4 = {0, 0, 0, 0};
  tdm_v8i z8 = {0, 0, 0, 0, 0, 0, 0, 0};
  __builtin_amdgcn_tensor_load_to_lds(g0, g1, z4, z4, z8, 0);
}

// ---------------------------------------------------------------------------
// Prep 1: x (f32) -> Xh (f16), elementwise, 4 elements/thread
// ---------------------------------------------------------------------------
__global__ __launch_bounds__(256) void conv_f32_f16(const float* __restrict__ src,
                                                    _Float16* __restrict__ dst) {
  size_t i = ((size_t)blockIdx.x * blockDim.x + threadIdx.x) * 4;
  float4 f = *(const float4*)(src + i);
  v4h o;
  o[0] = (_Float16)f.x; o[1] = (_Float16)f.y;
  o[2] = (_Float16)f.z; o[3] = (_Float16)f.w;
  *(v4h*)(dst + i) = o;
}

// ---------------------------------------------------------------------------
// Prep 2: W [K,N] f32 row-major -> Wt [N,K] f16 (LDS-tiled transpose)
// blockDim (32,8), grid (N/32, K/32)
// ---------------------------------------------------------------------------
__global__ __launch_bounds__(256) void transpose_f32_f16(const float* __restrict__ src,
                                                         _Float16* __restrict__ dst,
                                                         int K, int N) {
  __shared__ _Float16 t[32][33];
  const int tx = threadIdx.x;
  const int ty = threadIdx.y;
  const int n0 = blockIdx.x * 32;
  const int k0 = blockIdx.y * 32;
#pragma unroll
  for (int i = 0; i < 4; ++i) {
    int k = k0 + ty + 8 * i;
    t[tx][ty + 8 * i] = (_Float16)src[(size_t)k * N + n0 + tx];
  }
  __syncthreads();
#pragma unroll
  for (int i = 0; i < 4; ++i) {
    int n = n0 + ty + 8 * i;
    dst[(size_t)n * K + k0 + tx] = t[ty + 8 * i][tx];
  }
}

// ---------------------------------------------------------------------------
// Kernel 1: QKV = Xh @ Wqkv  (f16 operands, f32 accum).
// Block tile 128x64: 8 waves stacked in M share one B strip staged in LDS by
// the Tensor Data Mover, double-buffered one k-chunk ahead.
// Q,K stored [B,H,T,hd]; V stored transposed [B,H,hd,T].
// ---------------------------------------------------------------------------
__global__ __launch_bounds__(256) void qkv_gemm(const _Float16* __restrict__ Xh,
                                                const _Float16* __restrict__ Wt,  // [3072,1024]
                                                _Float16* __restrict__ Qb,
                                                _Float16* __restrict__ Kb,
                                                _Float16* __restrict__ Vt) {
  __shared__ _Float16 ldsB[2][64 * 32];   // two 4KB B-chunk buffers
  const int lane = threadIdx.x & 31;
  const int wave = threadIdx.x >> 5;
  const int hf   = lane >> 4;
  const int ln   = lane & 15;
  const int tn   = blockIdx.x % (D3_ / 64);   // 48 column strips
  const int tmb  = blockIdx.x / (D3_ / 64);   // 64 row blocks of 128
  const int mbase = tmb * 128 + wave * 16;
  const int row   = mbase + ln;

  const _Float16* arow  = Xh + (size_t)row * D_;
  const _Float16* Bbase = Wt + (size_t)(tn * 64) * D_;
  const unsigned  lb0 = (unsigned)(uintptr_t)&ldsB[0][0];
  const unsigned  lb1 = (unsigned)(uintptr_t)&ldsB[1][0];

  v8f cc[4] = {{}, {}, {}, {}};
  if (wave == 0) tdm_load_tile64x32(Bbase, lb0);
  for (int i = 0; i < D_ / 32; ++i) {
    const int k0 = i * 32;
    if (wave == 0) {
      if (i + 1 < D_ / 32) {
        tdm_load_tile64x32(Bbase + (size_t)(i + 1) * 32, (i & 1) ? lb0 : lb1);
        __builtin_amdgcn_s_wait_tensorcnt(1);   // chunk i landed; i+1 in flight
      } else {
        __builtin_amdgcn_s_wait_tensorcnt(0);
      }
    }
    __syncthreads();                            // B chunk i visible to all waves
    if (k0 + 64 < D_) __builtin_prefetch(arow + k0 + 64, 0, 1);
    const _Float16* bbuf = &ldsB[i & 1][0];
    v16h a = load_a_frag(arow, k0, hf);
#pragma unroll
    for (int j = 0; j < 4; ++j) {
      v16h bf = *(const v16h*)(bbuf + (j * 16 + ln) * 32 + 16 * hf);
      cc[j] = __builtin_amdgcn_wmma_f32_16x16x32_f16(false, a, false, bf,
                                                     (short)0, cc[j], false, false);
    }
    __syncthreads();                            // done reading before overwrite
  }

  // Column strip tn*64..tn*64+63 lies in exactly one of {Q,K,V} and one head.
  const int colbase = tn * 64;
  const int which   = colbase >> 10;            // 0=Q 1=K 2=V
  const int h       = (colbase & 1023) >> 6;
#pragma unroll
  for (int j = 0; j < 4; ++j) {
    const int hd = j * 16 + ln;
#pragma unroll
    for (int v = 0; v < 8; ++v) {
      int m  = mbase + v + 8 * hf;
      int bb = m >> 11;
      int t  = m & (T_ - 1);
      size_t bh = (size_t)(bb * H_ + h);
      _Float16 val = (_Float16)cc[j][v];
      if (which == 2)      Vt[(bh * HD_ + hd) * T_ + t] = val;   // transposed
      else if (which == 1) Kb[(bh * T_ + t) * HD_ + hd] = val;
      else                 Qb[(bh * T_ + t) * HD_ + hd] = val;
    }
  }
}

// ---------------------------------------------------------------------------
// Kernel 2: causal flash attention, one (b,h,16-query tile) per 1-wave block.
// K rows and V^T rows are contiguous -> all WMMA operands are vector loads.
// ---------------------------------------------------------------------------
__global__ __launch_bounds__(32) void attn(const _Float16* __restrict__ Qb,
                                           const _Float16* __restrict__ Kb,
                                           const _Float16* __restrict__ Vt,
                                           _Float16* __restrict__ Ob) {
  __shared__ _Float16 ldsP[16 * 32];   // P tile bounce (C-layout -> A-layout)

  const int lane = threadIdx.x & 31;
  const int hf   = lane >> 4;
  const int ln   = lane & 15;
  const int qt = blockIdx.x & 127;           // T/16 = 128 query tiles
  const int h  = (blockIdx.x >> 7) & 15;
  const int b  = blockIdx.x >> 11;
  const int q0 = qt * 16;

  const size_t bh = (size_t)(b * H_ + h);
  const _Float16* Q = Qb + bh * T_ * HD_;
  const _Float16* K = Kb + bh * T_ * HD_;
  const _Float16* Vrow[4];
#pragma unroll
  for (int j = 0; j < 4; ++j)
    Vrow[j] = Vt + (bh * HD_ + j * 16 + ln) * T_;

  const _Float16* qr = Q + (size_t)(q0 + ln) * HD_;
  v16h qa0 = load_a_frag(qr, 0, hf);
  v16h qa1 = load_a_frag(qr, 32, hf);

  v8f acc[4] = {{}, {}, {}, {}};
  float rowM[8], rowL[8];
#pragma unroll
  for (int v = 0; v < 8; ++v) { rowM[v] = -1e30f; rowL[v] = 0.f; }

  for (int key0 = 0; key0 < q0 + 16; key0 += 32) {
    // ---- S = Q @ K^T for 32 keys (two 16x16 tiles, hd reduced in 2 chunks)
    v8f s0 = {}, s1 = {};
    {
      const _Float16* kp0 = K + (size_t)(key0 + ln) * HD_ + 16 * hf;
      const _Float16* kp1 = K + (size_t)(key0 + 16 + ln) * HD_ + 16 * hf;
      v16h kf;
      kf = *(const v16h*)(kp0);
      s0 = __builtin_amdgcn_wmma_f32_16x16x32_f16(false, qa0, false, kf, (short)0, s0, false, false);
      kf = *(const v16h*)(kp0 + 32);
      s0 = __builtin_amdgcn_wmma_f32_16x16x32_f16(false, qa1, false, kf, (short)0, s0, false, false);
      kf = *(const v16h*)(kp1);
      s1 = __builtin_amdgcn_wmma_f32_16x16x32_f16(false, qa0, false, kf, (short)0, s1, false, false);
      kf = *(const v16h*)(kp1 + 32);
      s1 = __builtin_amdgcn_wmma_f32_16x16x32_f16(false, qa1, false, kf, (short)0, s1, false, false);
    }

    // ---- causal mask + online softmax (row m = v + 8*hf lives across the
    //      16 lanes of this half -> xor-shuffle reductions, masks 1,2,4,8)
    float p0[8], p1[8], alpha[8];
#pragma unroll
    for (int v = 0; v < 8; ++v) {
      int qi = q0 + v + 8 * hf;
      float a0 = (key0 + ln      <= qi) ? s0[v] * SCALE_ : -1e30f;
      float a1 = (key0 + 16 + ln <= qi) ? s1[v] * SCALE_ : -1e30f;
      float mx = fmaxf(a0, a1);
#pragma unroll
      for (int off = 1; off < 16; off <<= 1)
        mx = fmaxf(mx, __shfl_xor(mx, off, 32));
      float newM = fmaxf(rowM[v], mx);
      float al   = __expf(rowM[v] - newM);
      rowM[v]  = newM;
      alpha[v] = al;
      float e0 = __expf(a0 - newM);
      float e1 = __expf(a1 - newM);
      float rs = e0 + e1;
#pragma unroll
      for (int off = 1; off < 16; off <<= 1)
        rs += __shfl_xor(rs, off, 32);
      rowL[v] = rowL[v] * al + rs;
      p0[v] = e0;
      p1[v] = e1;
    }
#pragma unroll
    for (int j = 0; j < 4; ++j)
#pragma unroll
      for (int v = 0; v < 8; ++v) acc[j][v] *= alpha[v];

    // ---- repack P (C layout) into A layout via LDS
#pragma unroll
    for (int v = 0; v < 8; ++v) {
      int mrow = v + 8 * hf;
      ldsP[mrow * 32 + ln]      = (_Float16)p0[v];
      ldsP[mrow * 32 + 16 + ln] = (_Float16)p1[v];
    }
    __syncthreads();
    v16h pa;
#pragma unroll
    for (int e = 0; e < 16; ++e)
      pa[e] = ldsP[ln * 32 + a_k(e, hf)];
    __syncthreads();

    // ---- O += P @ V : V^T rows are contiguous -> vector loads
#pragma unroll
    for (int j = 0; j < 4; ++j) {
      v16h vf = *(const v16h*)(Vrow[j] + key0 + 16 * hf);
      acc[j] = __builtin_amdgcn_wmma_f32_16x16x32_f16(false, pa, false, vf,
                                                      (short)0, acc[j], false, false);
    }
  }

  // normalize and store O in [B,T,D] f16 (head h occupies cols h*64..h*64+63)
  float inv[8];
#pragma unroll
  for (int v = 0; v < 8; ++v) inv[v] = 1.0f / rowL[v];
  _Float16* Op = Ob + ((size_t)(b * T_) + q0) * D_ + h * HD_;
#pragma unroll
  for (int j = 0; j < 4; ++j)
#pragma unroll
    for (int v = 0; v < 8; ++v) {
      int mrow = v + 8 * hf;
      Op[(size_t)mrow * D_ + j * 16 + ln] = (_Float16)(acc[j][v] * inv[v]);
    }
}

// ---------------------------------------------------------------------------
// Kernel 3: out = O @ W_proj + b_proj (f16 operands, fp32 out).
// Same TDM-staged 128x64 block tiling as kernel 1.
// ---------------------------------------------------------------------------
__global__ __launch_bounds__(256) void proj_gemm(const _Float16* __restrict__ A,
                                                 const _Float16* __restrict__ Wt, // [1024,1024] col-major
                                                 const float* __restrict__ bias,
                                                 float* __restrict__ out) {
  __shared__ _Float16 ldsB[2][64 * 32];
  const int lane = threadIdx.x & 31;
  const int wave = threadIdx.x >> 5;
  const int hf   = lane >> 4;
  const int ln   = lane & 15;
  const int tn   = blockIdx.x % (D_ / 64);    // 16 column strips
  const int tmb  = blockIdx.x / (D_ / 64);
  const int mbase = tmb * 128 + wave * 16;
  const int row   = mbase + ln;

  const _Float16* arow  = A + (size_t)row * D_;
  const _Float16* Bbase = Wt + (size_t)(tn * 64) * D_;
  const unsigned  lb0 = (unsigned)(uintptr_t)&ldsB[0][0];
  const unsigned  lb1 = (unsigned)(uintptr_t)&ldsB[1][0];

  v8f cc[4] = {{}, {}, {}, {}};
  if (wave == 0) tdm_load_tile64x32(Bbase, lb0);
  for (int i = 0; i < D_ / 32; ++i) {
    const int k0 = i * 32;
    if (wave == 0) {
      if (i + 1 < D_ / 32) {
        tdm_load_tile64x32(Bbase + (size_t)(i + 1) * 32, (i & 1) ? lb0 : lb1);
        __builtin_amdgcn_s_wait_tensorcnt(1);
      } else {
        __builtin_amdgcn_s_wait_tensorcnt(0);
      }
    }
    __syncthreads();
    if (k0 + 64 < D_) __builtin_prefetch(arow + k0 + 64, 0, 1);
    const _Float16* bbuf = &ldsB[i & 1][0];
    v16h a = load_a_frag(arow, k0, hf);
#pragma unroll
    for (int j = 0; j < 4; ++j) {
      v16h bf = *(const v16h*)(bbuf + (j * 16 + ln) * 32 + 16 * hf);
      cc[j] = __builtin_amdgcn_wmma_f32_16x16x32_f16(false, a, false, bf,
                                                     (short)0, cc[j], false, false);
    }
    __syncthreads();
  }
#pragma unroll
  for (int j = 0; j < 4; ++j) {
    int col = tn * 64 + j * 16 + ln;
    float bv = bias[col];
#pragma unroll
    for (int v = 0; v < 8; ++v) {
      int m = mbase + v + 8 * hf;
      out[(size_t)m * D_ + col] = cc[j][v] + bv;
    }
  }
}

// ---------------------------------------------------------------------------
extern "C" void kernel_launch(void* const* d_in, const int* in_sizes, int n_in,
                              void* d_out, int out_size, void* d_ws, size_t ws_size,
                              hipStream_t stream) {
  const float* x     = (const float*)d_in[0];
  const float* Wqkv  = (const float*)d_in[1];
  const float* Wproj = (const float*)d_in[2];
  const float* bproj = (const float*)d_in[3];
  float* out = (float*)d_out;

  // Workspace carve-up (halves), ~88 MB total
  _Float16* Xh  = (_Float16*)d_ws;
  _Float16* Wqt = Xh  + (size_t)B_ * T_ * D_;
  _Float16* Wpt = Wqt + (size_t)D3_ * D_;
  _Float16* Qb  = Wpt + (size_t)D_ * D_;
  _Float16* Kb  = Qb  + (size_t)B_ * H_ * T_ * HD_;
  _Float16* Vt  = Kb  + (size_t)B_ * H_ * T_ * HD_;   // [B,H,hd,T]
  _Float16* Ob  = Vt  + (size_t)B_ * H_ * T_ * HD_;

  conv_f32_f16<<<(B_ * T_ * D_) / (256 * 4), 256, 0, stream>>>(x, Xh);
  transpose_f32_f16<<<dim3(D3_ / 32, D_ / 32), dim3(32, 8), 0, stream>>>(Wqkv, Wqt, D_, D3_);
  transpose_f32_f16<<<dim3(D_ / 32, D_ / 32), dim3(32, 8), 0, stream>>>(Wproj, Wpt, D_, D_);

  const int qkv_blocks  = ((B_ * T_) / 128) * (D3_ / 64);  // 64*48 = 3072
  const int attn_blocks = B_ * H_ * (T_ / 16);             // 8192
  const int proj_blocks = ((B_ * T_) / 128) * (D_ / 64);   // 64*16 = 1024

  qkv_gemm<<<qkv_blocks, 256, 0, stream>>>(Xh, Wqt, Qb, Kb, Vt);
  attn<<<attn_blocks, 32, 0, stream>>>(Qb, Kb, Vt, Ob);
  proj_gemm<<<proj_blocks, 256, 0, stream>>>(Ob, Wpt, bproj, out);
}
